// DotProductAttention_41626823032959
// MI455X (gfx1250) — matrix operand order repaired and checked by reading, not compiled, not run on hardware
//
#include <hip/hip_runtime.h>
#include <hip/hip_bf16.h>

typedef _Float16     v16h __attribute__((ext_vector_type(16)));
typedef __fp16       pk2  __attribute__((ext_vector_type(2)));   // cvt_pkrtz result type
typedef float        v8f  __attribute__((ext_vector_type(8)));
typedef unsigned int v4u  __attribute__((ext_vector_type(4)));

#define NB    16
#define LQ    2048
#define LK    2048
#define DD    64
#define CK    32                 // keys per chunk
#define NWAVE 4
#define QB    (NWAVE * 16)       // queries per block

// LDS layout (one buffer): K chunk f16 rows + V chunk key-pair-packed half2 dwords.
#define KROW_B   144                        // 64 halfs (128B) + 16B pad
#define K_BYTES  (CK * KROW_B)              // 4608
#define VROW_B   272                        // 64 dwords (256B) + 16B pad
#define VP_OFF   K_BYTES
#define SMEM_HALF (K_BYTES + 16 * VROW_B)   // 8960 B per buffer
#define SMEM_DB   (2 * SMEM_HALF)           // double buffered: 17920 B

// Workspace layout: f16 K [b][k][d] (128B rows), then packed V pairs [b][k/2][d] (256B rows)
#define K16_BYTES_G ((size_t)NB * LK * DD * 2)        // 4 MiB
#define VP_BYTES_G  ((size_t)NB * (LK / 2) * DD * 4)  // 4 MiB

// ---------------- CDNA5 async copy helpers (ASYNCcnt-tracked DMA to LDS) ----------------
__device__ __forceinline__ void async_b128_to_lds(unsigned lds_off, const void* gaddr) {
  asm volatile("global_load_async_to_lds_b128 %0, %1, off"
               :: "v"(lds_off), "v"(gaddr) : "memory");
}
__device__ __forceinline__ void wait_async0() {
  asm volatile("s_wait_asynccnt 0" ::: "memory");
}

// ---------------- prep: one-shot f32 -> f16 conversion into workspace ----------------
// K16[b][k][d] = (half)K ; VP[b][j][d] = pack(V[b][2j][d], V[b][2j+1][d])
__global__ __launch_bounds__(256) void
attn_prep_f16(const float* __restrict__ k, const float* __restrict__ v,
              unsigned int* __restrict__ k16, unsigned int* __restrict__ vp) {
  const int idx = blockIdx.x * 256 + threadIdx.x;   // 0 .. NB*LK*DD/2-1  (1,048,576)
  union { pk2 p; unsigned int u; } c;
  // K dword: row-major pairs along d
  {
    const int d2  = idx & (DD / 2 - 1);
    const int row = idx >> 5;                       // [0, NB*LK)
    const float* s = k + (size_t)row * DD + 2 * d2;
    c.p = __builtin_amdgcn_cvt_pkrtz(s[0], s[1]);
    k16[idx] = c.u;
  }
  // VP dword: pack the two keys of a pair at one d
  {
    const int d = idx & (DD - 1);
    const int j = (idx >> 6) & (LK / 2 - 1);
    const int b = idx >> 16;
    const float* s = v + ((size_t)b * LK + 2 * j) * DD + d;
    c.p = __builtin_amdgcn_cvt_pkrtz(s[0], s[DD]);
    vp[idx] = c.u;
  }
}

// ---------------- main: flash attention, async-DMA staged, double buffered ----------------
__global__ __launch_bounds__(NWAVE * 32) void
flash_attn_async(const float* __restrict__ q, const char* __restrict__ k16g,
                 const char* __restrict__ vpg, const int* __restrict__ vlen,
                 float* __restrict__ out) {
  __shared__ __align__(16) unsigned char smem[SMEM_DB];

  const int tid  = threadIdx.x;
  const int lane = tid & 31;
  const int wv   = tid >> 5;
  const int b    = blockIdx.x / (LQ / QB);
  const int qblk = blockIdx.x % (LQ / QB);
  const int q0   = qblk * QB + wv * 16;
  const int vl   = vlen[b];
  const int lq   = lane & 15;
  const int hi   = lane >> 4;
  const bool lo_half = (hi == 0);

  const float* qb_ = q + (size_t)b * LQ * DD;
  float*       ob_ = out + (size_t)b * LQ * DD;

  // ---- per-lane DMA addressing (LDS offset = low 32 bits of flat shared address) ----
  const unsigned lds_base = (unsigned)(uintptr_t)(void*)smem;
  // K: 8 lanes per 128B row -> 4 rows/instr, 2 instr/wave covers rows 8wv..8wv+7
  const unsigned kl0 = lds_base + (unsigned)((wv * 8 + (lane >> 3)) * KROW_B + (lane & 7) * 16);
  const unsigned kl1 = kl0 + 4 * KROW_B;
  const size_t   kgl0 = (size_t)(wv * 8 + (lane >> 3)) * 128 + (lane & 7) * 16;
  // VP: 16 lanes per 256B row -> 2 rows/instr, 2 instr/wave covers rows 4wv..4wv+3
  const unsigned pl0 = lds_base + VP_OFF + (unsigned)((wv * 4 + (lane >> 4)) * VROW_B + (lane & 15) * 16);
  const unsigned pl1 = pl0 + 2 * VROW_B;
  const size_t   pgl0 = (size_t)(wv * 4 + (lane >> 4)) * 256 + (lane & 15) * 16;

  const char* kgp = k16g + (size_t)b * LK * 128;        // advances 4096B per chunk
  const char* pgp = vpg + (size_t)b * (LK / 2) * 256;   // advances 4096B per chunk

  auto issue_chunk = [&](int bufsel) {
    const unsigned bo = bufsel ? (unsigned)SMEM_HALF : 0u;
    async_b128_to_lds(kl0 + bo, kgp + kgl0);
    async_b128_to_lds(kl1 + bo, kgp + kgl0 + 512);
    async_b128_to_lds(pl0 + bo, pgp + pgl0);
    async_b128_to_lds(pl1 + bo, pgp + pgl0 + 512);
    kgp += CK * 128;
    pgp += (CK / 2) * 256;
  };

  // ---- persistent Q^T B-operands (pre-scaled by 1/sqrt(64)) ----
  v16h Bq[2];
#pragma unroll
  for (int c = 0; c < 2; ++c) {
    const float* qp = qb_ + (size_t)(q0 + lq) * DD + c * 32 + hi * 16;
    union { v16h h; pk2 p[8]; } u;
#pragma unroll
    for (int j = 0; j < 8; ++j)
      u.p[j] = __builtin_amdgcn_cvt_pkrtz(qp[2 * j] * 0.125f, qp[2 * j + 1] * 0.125f);
    Bq[c] = u.h;
  }

  v8f acc[4] = {};
  float m  = -1e30f;
  float rs = 0.f;

  const int nchunk = (vl + CK - 1) / CK;   // exact: masked tail weights underflow to 0
  issue_chunk(0);                          // prologue DMA

  for (int ci = 0; ci < nchunk; ++ci) {
    wait_async0();        // this wave's copies for chunk ci are in LDS
    __syncthreads();      // everyone's copies done; previous compute on other buffer retired
    if (ci + 1 < nchunk) issue_chunk((ci + 1) & 1);   // overlap DMA with compute below

    const unsigned char* sb = smem + (ci & 1) * SMEM_HALF;
    const int kbase = ci * CK;

    // ---- S^T = K * Q^T : two 16-key tiles, contraction over D=64 ----
    v8f st[2];
#pragma unroll
    for (int t = 0; t < 2; ++t) {
      v8f c8 = {};
#pragma unroll
      for (int c = 0; c < 2; ++c) {
        const unsigned char* rb = &sb[(t * 16 + lq) * KROW_B + c * 64 + hi * 32];
        union { v16h h; v4u u[2]; } a;
        a.u[0] = *(const v4u*)rb;
        a.u[1] = *(const v4u*)(rb + 16);
        c8 = __builtin_amdgcn_wmma_f32_16x16x32_f16(false, a.h, false, Bq[c],
                                                    (short)0, c8, false, false);
      }
      st[t] = c8;
    }

    // ---- key masking (boundary chunk only; uniform branch) ----
    if (kbase + CK > vl) {
#pragma unroll
      for (int r = 0; r < 8; ++r) {
        const int k0 = kbase + hi * 8 + r;
        st[0][r] = (k0 < vl)      ? st[0][r] : -1e6f;
        st[1][r] = (k0 + 16 < vl) ? st[1][r] : -1e6f;
      }
    }

    // ---- online softmax (keys in-lane across VGPRs + one half-swap) ----
    float cm = st[0][0];
#pragma unroll
    for (int r = 0; r < 8; ++r) { cm = fmaxf(cm, st[0][r]); cm = fmaxf(cm, st[1][r]); }
    cm = fmaxf(cm, __shfl_xor(cm, 16));
    const float mn = fmaxf(m, cm);

    if (__any(cm > m)) {       // max changed for some query: rescale state (rare after warmup)
      const float sc = __expf(m - mn);
      rs *= sc;
#pragma unroll
      for (int r = 0; r < 8; ++r) {
        const float sl = __shfl(sc, r);
        const float shh = __shfl(sc, r + 8);
        const float sr = lo_half ? sl : shh;
#pragma unroll
        for (int dt = 0; dt < 4; ++dt) acc[dt][r] *= sr;
      }
    }
    m = mn;

    float p0[8], p1[8], sh = 0.f;
#pragma unroll
    for (int r = 0; r < 8; ++r) {
      p0[r] = __expf(st[0][r] - mn);
      p1[r] = __expf(st[1][r] - mn);
      sh += p0[r] + p1[r];
    }
    sh += __shfl_xor(sh, 16);
    rs += sh;

    // ---- P: S^T D-layout -> A-layout of P is pure in-lane packing ----
    union { v16h h; pk2 p[8]; } ap;
#pragma unroll
    for (int j2 = 0; j2 < 4; ++j2) {
      ap.p[j2]     = __builtin_amdgcn_cvt_pkrtz(p0[2 * j2], p0[2 * j2 + 1]);
      ap.p[4 + j2] = __builtin_amdgcn_cvt_pkrtz(p1[2 * j2], p1[2 * j2 + 1]);
    }

    // ---- O += P * V (4 d-tiles of 16) ----
#pragma unroll
    for (int dt = 0; dt < 4; ++dt) {
      union { v16h h; unsigned int u[8]; } bv;
#pragma unroll
      for (int j2 = 0; j2 < 8; ++j2)
        bv.u[j2] = *(const unsigned int*)
            &sb[VP_OFF + ((j2 >> 2) * 8 + hi * 4 + (j2 & 3)) * VROW_B + (dt * 16 + lq) * 4];
      acc[dt] = __builtin_amdgcn_wmma_f32_16x16x32_f16(false, ap.h, false, bv.h,
                                                       (short)0, acc[dt], false, false);
    }
  }

  // ---- epilogue: normalize + store ----
  const float inv = 1.f / rs;
#pragma unroll
  for (int r = 0; r < 8; ++r) {
    const float il = __shfl(inv, r);
    const float ih = __shfl(inv, r + 8);
    const float ir = lo_half ? il : ih;
    const int row = q0 + hi * 8 + r;
#pragma unroll
    for (int dt = 0; dt < 4; ++dt)
      ob_[(size_t)row * DD + dt * 16 + lq] = acc[dt][r] * ir;
  }
}

// ---------------- fallback (no workspace): round-2 kernel, in-loop conversion ----------------
__global__ __launch_bounds__(NWAVE * 32) void
flash_attn_f16wmma(const float* __restrict__ q, const float* __restrict__ kk,
                   const float* __restrict__ vv, const int* __restrict__ vlen,
                   float* __restrict__ out) {
  __shared__ __align__(16) unsigned char smem[SMEM_HALF];

  const int tid  = threadIdx.x;
  const int lane = tid & 31;
  const int wv   = tid >> 5;
  const int b    = blockIdx.x / (LQ / QB);
  const int qblk = blockIdx.x % (LQ / QB);
  const int q0   = qblk * QB + wv * 16;
  const int vl   = vlen[b];
  const int lq   = lane & 15;
  const int hi   = lane >> 4;
  const bool lo_half = (hi == 0);

  const float* qb_ = q  + (size_t)b * LQ * DD;
  const float* kb_ = kk + (size_t)b * LK * DD;
  const float* vb_ = vv + (size_t)b * LK * DD;
  float*       ob_ = out + (size_t)b * LQ * DD;

  v16h Bq[2];
#pragma unroll
  for (int c = 0; c < 2; ++c) {
    const float* qp = qb_ + (size_t)(q0 + lq) * DD + c * 32 + hi * 16;
    union { v16h h; pk2 p[8]; } u;
#pragma unroll
    for (int j = 0; j < 8; ++j)
      u.p[j] = __builtin_amdgcn_cvt_pkrtz(qp[2 * j] * 0.125f, qp[2 * j + 1] * 0.125f);
    Bq[c] = u.h;
  }

  v8f acc[4] = {};
  float m  = -1e30f;
  float rs = 0.f;
  const int nchunk = (vl + CK - 1) / CK;

  for (int ci = 0; ci < nchunk; ++ci) {
    const int kbase = ci * CK;
    {
      const int r  = tid >> 2;
      const int cb = (tid & 3) * 16;
      const float* kp = kb_ + (size_t)(kbase + r) * DD + cb;
      if (kbase + CK < LK) __builtin_prefetch(kp + CK * DD, 0, 3);
      union { v4u u[2]; pk2 p[8]; } w;
#pragma unroll
      for (int j = 0; j < 8; ++j)
        w.p[j] = __builtin_amdgcn_cvt_pkrtz(kp[2 * j], kp[2 * j + 1]);
      v4u* dst = (v4u*)&smem[r * KROW_B + cb * 2];
      dst[0] = w.u[0]; dst[1] = w.u[1];
    }
    {
      const int j  = tid >> 3;
      const int db = (tid & 7) * 8;
      const float* vp0 = vb_ + (size_t)(kbase + 2 * j) * DD + db;
      const float* vp1 = vp0 + DD;
      if (kbase + CK < LK) __builtin_prefetch(vp0 + CK * DD, 0, 3);
      union { v4u u[2]; pk2 p[8]; } w;
#pragma unroll
      for (int e = 0; e < 8; ++e)
        w.p[e] = __builtin_amdgcn_cvt_pkrtz(vp0[e], vp1[e]);
      v4u* dst = (v4u*)&smem[VP_OFF + j * VROW_B + db * 4];
      dst[0] = w.u[0]; dst[1] = w.u[1];
    }
    __syncthreads();

    v8f st[2];
#pragma unroll
    for (int t = 0; t < 2; ++t) {
      v8f c8 = {};
#pragma unroll
      for (int c = 0; c < 2; ++c) {
        const unsigned char* rb = &smem[(t * 16 + lq) * KROW_B + c * 64 + hi * 32];
        union { v16h h; v4u u[2]; } a;
        a.u[0] = *(const v4u*)rb;
        a.u[1] = *(const v4u*)(rb + 16);
        c8 = __builtin_amdgcn_wmma_f32_16x16x32_f16(false, a.h, false, Bq[c],
                                                    (short)0, c8, false, false);
      }
      st[t] = c8;
    }

    if (kbase + CK > vl) {
#pragma unroll
      for (int r = 0; r < 8; ++r) {
        const int k0 = kbase + hi * 8 + r;
        st[0][r] = (k0 < vl)      ? st[0][r] : -1e6f;
        st[1][r] = (k0 + 16 < vl) ? st[1][r] : -1e6f;
      }
    }

    float cm = st[0][0];
#pragma unroll
    for (int r = 0; r < 8; ++r) { cm = fmaxf(cm, st[0][r]); cm = fmaxf(cm, st[1][r]); }
    cm = fmaxf(cm, __shfl_xor(cm, 16));
    const float mn = fmaxf(m, cm);
    if (__any(cm > m)) {
      const float sc = __expf(m - mn);
      rs *= sc;
#pragma unroll
      for (int r = 0; r < 8; ++r) {
        const float sl = __shfl(sc, r);
        const float shh = __shfl(sc, r + 8);
        const float sr = lo_half ? sl : shh;
#pragma unroll
        for (int dt = 0; dt < 4; ++dt) acc[dt][r] *= sr;
      }
    }
    m = mn;

    float p0[8], p1[8], sh = 0.f;
#pragma unroll
    for (int r = 0; r < 8; ++r) {
      p0[r] = __expf(st[0][r] - mn);
      p1[r] = __expf(st[1][r] - mn);
      sh += p0[r] + p1[r];
    }
    sh += __shfl_xor(sh, 16);
    rs += sh;

    union { v16h h; pk2 p[8]; } ap;
#pragma unroll
    for (int j2 = 0; j2 < 4; ++j2) {
      ap.p[j2]     = __builtin_amdgcn_cvt_pkrtz(p0[2 * j2], p0[2 * j2 + 1]);
      ap.p[4 + j2] = __builtin_amdgcn_cvt_pkrtz(p1[2 * j2], p1[2 * j2 + 1]);
    }

#pragma unroll
    for (int dt = 0; dt < 4; ++dt) {
      union { v16h h; unsigned int u[8]; } bv;
#pragma unroll
      for (int j2 = 0; j2 < 8; ++j2)
        bv.u[j2] = *(const unsigned int*)
            &smem[VP_OFF + ((j2 >> 2) * 8 + hi * 4 + (j2 & 3)) * VROW_B + (dt * 16 + lq) * 4];
      acc[dt] = __builtin_amdgcn_wmma_f32_16x16x32_f16(false, ap.h, false, bv.h,
                                                       (short)0, acc[dt], false, false);
    }
    __syncthreads();
  }

  const float inv = 1.f / rs;
#pragma unroll
  for (int r = 0; r < 8; ++r) {
    const float il = __shfl(inv, r);
    const float ih = __shfl(inv, r + 8);
    const float ir = lo_half ? il : ih;
    const int row = q0 + hi * 8 + r;
#pragma unroll
    for (int dt = 0; dt < 4; ++dt)
      ob_[(size_t)row * DD + dt * 16 + lq] = acc[dt][r] * ir;
  }
}

extern "C" void kernel_launch(void* const* d_in, const int* in_sizes, int n_in,
                              void* d_out, int out_size, void* d_ws, size_t ws_size,
                              hipStream_t stream) {
  (void)in_sizes; (void)n_in; (void)out_size;
  const float* q    = (const float*)d_in[0];
  const float* k    = (const float*)d_in[1];
  const float* v    = (const float*)d_in[2];
  const int*   vlen = (const int*)d_in[3];
  float* out = (float*)d_out;

  dim3 grid(NB * (LQ / QB));   // 512 blocks
  dim3 block(NWAVE * 32);      // 4 waves

  if (d_ws && ws_size >= K16_BYTES_G + VP_BYTES_G) {
    unsigned int* k16 = (unsigned int*)d_ws;
    unsigned int* vp  = (unsigned int*)((char*)d_ws + K16_BYTES_G);
    attn_prep_f16<<<dim3((NB * LK * DD / 2) / 256), dim3(256), 0, stream>>>(k, v, k16, vp);
    flash_attn_async<<<grid, block, 0, stream>>>(q, (const char*)k16, (const char*)vp,
                                                 vlen, out);
  } else {
    flash_attn_f16wmma<<<grid, block, 0, stream>>>(q, k, v, vlen, out);
  }
}
